// CauseAttention_3152505995878
// MI455X (gfx1250) — matrix operand order repaired and checked by reading, not compiled, not run on hardware
//
#include <hip/hip_runtime.h>

typedef unsigned short u16;
typedef unsigned int   u32;
typedef unsigned long long u64;

typedef __attribute__((ext_vector_type(16))) __bf16 v16bf;
typedef __attribute__((ext_vector_type(8)))  float  v8f;
typedef __attribute__((ext_vector_type(4)))  unsigned int v4u;
typedef __attribute__((ext_vector_type(8)))  int v8i_t;
typedef __attribute__((ext_vector_type(4)))  int v4i_t;

union FragAB { v16bf v; uint4 q[2]; u16 s[16]; };
union FragC  { v8f v; float f[8]; };

__device__ __forceinline__ u16 f2bf(float f) {
    u32 x = __float_as_uint(f);
    u32 r = x + 0x7FFFu + ((x >> 16) & 1u);
    return (u16)(r >> 16);
}

#define Bsz 16
#define Tn  1024
#define Nn  1024
#define Hn  16
#define HDm 64
#define BT  (Bsz * Tn)      /* 16384 */
#define N3  (3 * Nn)        /* 3072  */
#define ATT_SCALE 0.03125f  /* 1/sqrt(1024) */

// ---------------------------------------------------------------------------
// TDM: async-load a [128 rows x 32 bf16] tile (row stride K elems) into LDS,
// with LDS padding 64B rows -> 80B rows (pad_interval=16 dwords, amount=4 dw).
// Issued by one wave; completion via s_wait_tensorcnt + block barrier.
// ---------------------------------------------------------------------------
__device__ __forceinline__ void tdm_load_tile_A(const u16* gsrc, u32 lds_off, int K) {
    u64 ga = (u64)(size_t)gsrc;
    v4u g0;
    g0[0] = 1u;                                   // count=1, user mode
    g0[1] = lds_off;                              // lds_addr (bytes)
    g0[2] = (u32)(ga & 0xffffffffu);              // global_addr[31:0]
    g0[3] = (u32)((ga >> 32) & 0x01ffffffu)       // global_addr[56:32]
            | 0x80000000u;                        // type = 2 ("image")
    v8i_t g1;
    g1[0] = (int)((1u << 16)      // data_size = 1 -> 2 bytes
                | (1u << 20)      // pad_enable
                | (3u << 22)      // pad_interval: 16 dwords (one 64B row)
                | (3u << 25));    // pad_amount: 4 dwords (16B)
    g1[1] = (int)((u32)K << 16);                  // tensor_dim0[15:0] @ bits 63:48
    g1[2] = (int)((u32)(BT & 0xffff) << 16);      // tensor_dim1[15:0] @ bits 95:80
    g1[3] = (int)(32u << 16);                     // tile_dim0 = 32 @ bits 127:112
    g1[4] = 128;                                  // tile_dim1 = 128 @ bits 143:128
    g1[5] = K;                                    // tensor_dim0_stride[31:0]
    g1[6] = 0;
    g1[7] = 0;
    v4i_t z4 = {0, 0, 0, 0};
#if __clang_major__ >= 23
    v8i_t z8 = {0, 0, 0, 0, 0, 0, 0, 0};
    __builtin_amdgcn_tensor_load_to_lds(g0, g1, z4, z4, z8, 0);
#else
    __builtin_amdgcn_tensor_load_to_lds(g0, g1, z4, z4, 0);
#endif
}

// ---------------------------------------------------------------------------
// fp32 -> bf16 conversion (vectorized, n multiple of 4)
// ---------------------------------------------------------------------------
__global__ __launch_bounds__(256)
void cvt_f32_to_bf16(const float* __restrict__ src, u16* __restrict__ dst, int n4) {
    int stride = gridDim.x * blockDim.x;
    for (int i = blockIdx.x * blockDim.x + threadIdx.x; i < n4; i += stride) {
        float4 f = ((const float4*)src)[i];
        uint2 o;
        o.x = (u32)f2bf(f.x) | ((u32)f2bf(f.y) << 16);
        o.y = (u32)f2bf(f.z) | ((u32)f2bf(f.w) << 16);
        ((uint2*)dst)[i] = o;
    }
}

// ---------------------------------------------------------------------------
// bf16 WMMA GEMM:  C[M,N] = A[M,K] * B[K,N] + bias
// Block tile 128x128, BK=32, 256 threads = 8 waves, each wave 32(M)x64(N).
// A tile staged by TDM (async tensor DMA); B tile staged transposed by VALU.
// F32OUT selects fp32 vs bf16 output at compile time.
// ---------------------------------------------------------------------------
template <bool F32OUT>
__global__ __launch_bounds__(256)
void gemm_bf16_kernel(const u16* __restrict__ A, const u16* __restrict__ Bm,
                      const float* __restrict__ bias, void* __restrict__ outp,
                      int M, int N, int K)
{
    __shared__ __align__(16) u16 As[128][40];  // row-major [m][k], pad 32->40
    __shared__ __align__(16) u16 Bs[128][40];  // transposed [n][k], pad

    const int tid  = threadIdx.x;
    const int lane = tid & 31;
    const int wave = tid >> 5;
    const int wm   = (wave & 3) * 32;   // wave M base within tile
    const int wn   = (wave >> 2) * 64;  // wave N base within tile
    const int half = lane >> 4;
    const int lr   = lane & 15;

    const int mblock = blockIdx.y * 128;
    const int nblock = blockIdx.x * 128;
    const u32 As_off = (u32)(size_t)&As[0][0];  // LDS byte address for TDM

    FragC c[2][4];
#pragma unroll
    for (int mi = 0; mi < 2; ++mi)
#pragma unroll
        for (int ni = 0; ni < 4; ++ni)
#pragma unroll
            for (int v = 0; v < 8; ++v) c[mi][ni].f[v] = 0.0f;

    const int kr = tid >> 3;          // 0..31  (B stage row)
    const int nb = (tid & 7) * 16;    // 0..112 (B stage col group)

    for (int k0 = 0; k0 < K; k0 += 32) {
        __syncthreads();
        // ---- stage A tile via Tensor Data Mover (one wave issues DMA) ----
        if (wave == 0)
            tdm_load_tile_A(A + (size_t)mblock * K + k0, As_off, K);

        // ---- stage B tile transposed: 32 k-rows x 128 n, 16 elems/thread ----
        {
            const u16* src = Bm + (size_t)(k0 + kr) * N + nblock + nb;
            uint4 p0 = *(const uint4*)(src);
            uint4 p1 = *(const uint4*)(src + 8);
            if (k0 + 32 < K)
                __builtin_prefetch(src + 32, 0, 1);   // same k row, 32 elems further along n
            u16 tmp[16];
            *(uint4*)&tmp[0] = p0;
            *(uint4*)&tmp[8] = p1;
#pragma unroll
            for (int j = 0; j < 16; ++j) Bs[nb + j][kr] = tmp[j];
        }
        if (wave == 0)
            __builtin_amdgcn_s_wait_tensorcnt(0);
        __syncthreads();

        FragAB a[2], b[4];
#pragma unroll
        for (int mi = 0; mi < 2; ++mi) {
            const u16* rp = &As[wm + mi * 16 + lr][0];
            a[mi].q[0] = *(const uint4*)(rp + half * 8);        // K 0..7  / 8..15
            a[mi].q[1] = *(const uint4*)(rp + 16 + half * 8);   // K 16..23 / 24..31
        }
#pragma unroll
        for (int ni = 0; ni < 4; ++ni) {
            const u16* rp = &Bs[wn + ni * 16 + lr][0];
            b[ni].q[0] = *(const uint4*)(rp + half * 8);        // K 0..7  / 8..15
            b[ni].q[1] = *(const uint4*)(rp + 16 + half * 8);   // K 16..23 / 24..31
        }
#pragma unroll
        for (int mi = 0; mi < 2; ++mi)
#pragma unroll
            for (int ni = 0; ni < 4; ++ni)
                c[mi][ni].v = __builtin_amdgcn_wmma_f32_16x16x32_bf16(
                    false, a[mi].v, false, b[ni].v, (short)0, c[mi][ni].v, false, false);
    }

    // ---- epilogue: bias + store (compile-time output mode, no branches) ----
#pragma unroll
    for (int mi = 0; mi < 2; ++mi) {
#pragma unroll
        for (int ni = 0; ni < 4; ++ni) {
            int n = nblock + wn + ni * 16 + lr;
            float bv = bias[n];
            size_t base = (size_t)(mblock + wm + mi * 16 + half * 8) * (size_t)N + n;
#pragma unroll
            for (int v = 0; v < 8; ++v) {
                float val = c[mi][ni].f[v] + bv;
                if (F32OUT)
                    ((float*)outp)[base + (size_t)v * N] = val;
                else
                    ((u16*)outp)[base + (size_t)v * N] = f2bf(val);
            }
        }
    }
}

// ---------------------------------------------------------------------------
// Flash attention: one block = one (b,h) x 128-row Q tile.
// 256 threads = 8 waves arranged 4(M) x 2(N). K-tiles of 64 keys.
// ---------------------------------------------------------------------------
__global__ __launch_bounds__(256)
void flash_attn_kernel(const u16* __restrict__ qkv, u16* __restrict__ yb)
{
    __shared__ __align__(16) u16 Qs[128][72];   // [qrow][hd], pad 64->72
    __shared__ __align__(16) u16 Ps[128][72];   // [qrow][key], pad
    __shared__ __align__(16) u16 Vt[64][72];    // transposed [hd][key], pad
    __shared__ float mrow[128], lrow[128], rfac[128];
    __shared__ float partmax[128][2], partsum[128][2];

    const int tid  = threadIdx.x;
    const int lane = tid & 31;
    const int wave = tid >> 5;
    const int wm   = (wave & 3) * 32;   // q-row base within 128
    const int wn   = (wave >> 2) * 32;  // col base (keys for S, hd for O)
    const int wcol = wave >> 2;         // 0/1 column-group id
    const int half = lane >> 4;
    const int lr   = lane & 15;

    const int qb = blockIdx.x * 128;
    const int h  = blockIdx.y;
    const int b  = blockIdx.z;

    // stage Q tile
    {
        int r  = tid >> 1;
        int cb = (tid & 1) * 32;
        const u16* src = qkv + (size_t)(b * Tn + qb + r) * N3 + h * HDm + cb;
#pragma unroll
        for (int j = 0; j < 4; ++j)
            *(uint4*)&Qs[r][cb + j * 8] = *(const uint4*)(src + j * 8);
    }
    if (tid < 128) { mrow[tid] = -1e30f; lrow[tid] = 0.0f; }

    FragC o[2][2];
#pragma unroll
    for (int mi = 0; mi < 2; ++mi)
#pragma unroll
        for (int ni = 0; ni < 2; ++ni)
#pragma unroll
            for (int v = 0; v < 8; ++v) o[mi][ni].f[v] = 0.0f;

    for (int kt = 0; kt < Tn / 64; ++kt) {
        const int kb = kt * 64;
        __syncthreads();  // protect Vt/Ps from previous iteration's readers

        {   // stage V tile transposed: Vt[hd][key]
            int vr = tid >> 2;           // key row 0..63
            int cb = (tid & 3) * 16;     // hd col group
            const u16* src = qkv + (size_t)(b * Tn + kb + vr) * N3 + 2 * Nn + h * HDm + cb;
            u16 tmp[16];
            *(uint4*)&tmp[0] = *(const uint4*)(src);
            *(uint4*)&tmp[8] = *(const uint4*)(src + 8);
#pragma unroll
            for (int j = 0; j < 16; ++j) Vt[cb + j][vr] = tmp[j];
        }

        // ---- S = scale * Q K^T (wave computes 32x32) ----
        FragC sfr[2][2];
#pragma unroll
        for (int mi = 0; mi < 2; ++mi)
#pragma unroll
            for (int ni = 0; ni < 2; ++ni)
#pragma unroll
                for (int v = 0; v < 8; ++v) sfr[mi][ni].f[v] = 0.0f;

#pragma unroll
        for (int s = 0; s < 2; ++s) {   // hd = 64 => two K=32 steps
            FragAB a[2], kb_f[2];
#pragma unroll
            for (int mi = 0; mi < 2; ++mi) {
                const u16* rp = &Qs[wm + mi * 16 + lr][s * 32];
                a[mi].q[0] = *(const uint4*)(rp + half * 8);
                a[mi].q[1] = *(const uint4*)(rp + 16 + half * 8);
            }
#pragma unroll
            for (int ni = 0; ni < 2; ++ni) {
                int key = kb + wn + ni * 16 + lr;
                const u16* kp = qkv + (size_t)(b * Tn + key) * N3 + Nn + h * HDm
                                + s * 32 + half * 8;
                kb_f[ni].q[0] = *(const uint4*)(kp);
                kb_f[ni].q[1] = *(const uint4*)(kp + 16);
            }
#pragma unroll
            for (int mi = 0; mi < 2; ++mi)
#pragma unroll
                for (int ni = 0; ni < 2; ++ni)
                    sfr[mi][ni].v = __builtin_amdgcn_wmma_f32_16x16x32_bf16(
                        false, a[mi].v, false, kb_f[ni].v, (short)0, sfr[mi][ni].v,
                        false, false);
        }
#pragma unroll
        for (int mi = 0; mi < 2; ++mi)
#pragma unroll
            for (int ni = 0; ni < 2; ++ni)
#pragma unroll
                for (int v = 0; v < 8; ++v) sfr[mi][ni].f[v] *= ATT_SCALE;

        // ---- per-row max partials (reduce across ni, then 16-lane group) ----
#pragma unroll
        for (int mi = 0; mi < 2; ++mi) {
            float mx[8];
#pragma unroll
            for (int v = 0; v < 8; ++v)
                mx[v] = fmaxf(sfr[mi][0].f[v], sfr[mi][1].f[v]);
#pragma unroll
            for (int off = 1; off < 16; off <<= 1)
#pragma unroll
                for (int v = 0; v < 8; ++v)
                    mx[v] = fmaxf(mx[v], __shfl_xor(mx[v], off));
            if (lr == 0)
#pragma unroll
                for (int v = 0; v < 8; ++v)
                    partmax[wm + mi * 16 + half * 8 + v][wcol] = mx[v];
        }
        __syncthreads();

        if (tid < 128) {
            float pm   = fmaxf(partmax[tid][0], partmax[tid][1]);
            float mold = mrow[tid];
            float mnew = fmaxf(mold, pm);
            mrow[tid]  = mnew;
            rfac[tid]  = __expf(mold - mnew);
        }
        __syncthreads();

        // ---- P = exp(S - m), row-sum partials, Ps <- bf16(P) ----
#pragma unroll
        for (int mi = 0; mi < 2; ++mi) {
            float sm[8];
#pragma unroll
            for (int v = 0; v < 8; ++v) {
                int row  = wm + mi * 16 + half * 8 + v;
                float mn = mrow[row];
                float acc = 0.0f;
#pragma unroll
                for (int ni = 0; ni < 2; ++ni) {
                    float p = __expf(sfr[mi][ni].f[v] - mn);
                    Ps[row][wn + ni * 16 + lr] = f2bf(p);
                    acc += p;
                }
                sm[v] = acc;
            }
#pragma unroll
            for (int off = 1; off < 16; off <<= 1)
#pragma unroll
                for (int v = 0; v < 8; ++v)
                    sm[v] += __shfl_xor(sm[v], off);
            if (lr == 0)
#pragma unroll
                for (int v = 0; v < 8; ++v)
                    partsum[wm + mi * 16 + half * 8 + v][wcol] = sm[v];
        }
        __syncthreads();

        if (tid < 128)
            lrow[tid] = lrow[tid] * rfac[tid] + partsum[tid][0] + partsum[tid][1];

        // ---- O = O * rfac + P x V ----
#pragma unroll
        for (int mi = 0; mi < 2; ++mi)
#pragma unroll
            for (int ni = 0; ni < 2; ++ni)
#pragma unroll
                for (int v = 0; v < 8; ++v)
                    o[mi][ni].f[v] *= rfac[wm + mi * 16 + half * 8 + v];

#pragma unroll
        for (int s = 0; s < 2; ++s) {   // keys 64 => two K=32 steps
            FragAB pa[2], pb[2];
#pragma unroll
            for (int mi = 0; mi < 2; ++mi) {
                const u16* rp = &Ps[wm + mi * 16 + lr][s * 32];
                pa[mi].q[0] = *(const uint4*)(rp + half * 8);
                pa[mi].q[1] = *(const uint4*)(rp + 16 + half * 8);
            }
#pragma unroll
            for (int ni = 0; ni < 2; ++ni) {
                const u16* rp = &Vt[wn + ni * 16 + lr][s * 32];
                pb[ni].q[0] = *(const uint4*)(rp + half * 8);
                pb[ni].q[1] = *(const uint4*)(rp + 16 + half * 8);
            }
#pragma unroll
            for (int mi = 0; mi < 2; ++mi)
#pragma unroll
                for (int ni = 0; ni < 2; ++ni)
                    o[mi][ni].v = __builtin_amdgcn_wmma_f32_16x16x32_bf16(
                        false, pa[mi].v, false, pb[ni].v, (short)0, o[mi][ni].v,
                        false, false);
        }
    }
    __syncthreads();

    // ---- normalize and store y (bf16, [B*T, N] with head offset) ----
#pragma unroll
    for (int mi = 0; mi < 2; ++mi)
#pragma unroll
        for (int ni = 0; ni < 2; ++ni)
#pragma unroll
            for (int v = 0; v < 8; ++v) {
                int row = wm + mi * 16 + half * 8 + v;
                int col = wn + ni * 16 + lr;
                float val = o[mi][ni].f[v] / lrow[row];
                yb[(size_t)(b * Tn + qb + row) * Nn + h * HDm + col] = f2bf(val);
            }
}

// ---------------------------------------------------------------------------
// launch
// ---------------------------------------------------------------------------
extern "C" void kernel_launch(void* const* d_in, const int* in_sizes, int n_in,
                              void* d_out, int out_size, void* d_ws, size_t ws_size,
                              hipStream_t stream) {
    const float* x      = (const float*)d_in[0];   // [B,T,N]
    const float* W_attn = (const float*)d_in[1];   // [N,3N]
    const float* b_attn = (const float*)d_in[2];   // [3N]
    const float* W_proj = (const float*)d_in[3];   // [N,N]
    const float* b_proj = (const float*)d_in[4];   // [N]
    float* out = (float*)d_out;                    // [B,T,N]

    u16* xb   = (u16*)d_ws;                        // B*T*N
    u16* Wab  = xb   + (size_t)BT * Nn;            // N*3N
    u16* Wpb  = Wab  + (size_t)Nn * N3;            // N*N
    u16* qkvb = Wpb  + (size_t)Nn * Nn;            // B*T*3N
    u16* yb   = qkvb + (size_t)BT * N3;            // B*T*N

    cvt_f32_to_bf16<<<4096, 256, 0, stream>>>(x,      xb,  (BT * Nn) / 4);
    cvt_f32_to_bf16<<<2048, 256, 0, stream>>>(W_attn, Wab, (Nn * N3) / 4);
    cvt_f32_to_bf16<<<1024, 256, 0, stream>>>(W_proj, Wpb, (Nn * Nn) / 4);

    // qkv = x @ W_attn + b_attn  (bf16 out)
    gemm_bf16_kernel<false><<<dim3(N3 / 128, BT / 128), 256, 0, stream>>>(
        xb, Wab, b_attn, qkvb, BT, N3, Nn);

    // y = softmax(q k^T * scale) v  (bf16 out)
    flash_attn_kernel<<<dim3(Tn / 128, Hn, Bsz), 256, 0, stream>>>(qkvb, yb);

    // out = y @ W_proj + b_proj  (fp32 out)
    gemm_bf16_kernel<true><<<dim3(Nn / 128, BT / 128), 256, 0, stream>>>(
        yb, Wpb, b_proj, out, BT, Nn, Nn);
}
